// GRUTridentDecoder_2362232013338
// MI455X (gfx1250) — matrix operand order repaired and checked by reading, compiled, not run on hardware
//
#include <hip/hip_runtime.h>

// ---------------------------------------------------------------------------
// GRUTridentDecoder for MI455X (gfx1250, wave32, WMMA bf16 16x16x32, f32 acc)
// B=32, H=256, V=512, A=3, NANN=8, DEPTH=7, N=3280 nodes
// Weights pre-converted to bf16; all WMMA fragments loaded as aligned vector
// loads (2x b128 per fragment) to avoid scalar-load/merge codegen.
// ---------------------------------------------------------------------------

typedef __attribute__((ext_vector_type(16))) __bf16 v16bf;
typedef __attribute__((ext_vector_type(8)))  __bf16 v8bf;
typedef __attribute__((ext_vector_type(8)))  float  v8f;

#define Bc     32
#define Hc     256
#define Vc     512
#define Gc     768      // 3*H
#define AGc    2304     // A*3*H
#define NNODES 3280

__device__ __forceinline__ float sigmoidf_(float x) {
  return 1.0f / (1.0f + __expf(-x));
}

// D = A*B + C, bf16 inputs, f32 accumulate (codegen-confirmed builtin)
__device__ __forceinline__ v8f wmma_bf16(v16bf a, v16bf b, v8f c) {
  return __builtin_amdgcn_wmma_f32_16x16x32_bf16(false, a, false, b,
                                                 (short)0, c, false, false);
}

// A fragment (16x32, bf16) from an LDS row pointer at k0.
// lanes 0-15: M=lane, K = e (e<8) / 16+(e-8); lanes 16-31: +8 on K.
// Two aligned 16B vector loads + register shuffle.
__device__ __forceinline__ v16bf load_afrag_lds(const __bf16* row, int kb /*hi*8*/) {
  v8bf lo = *(const v8bf*)(row + kb);
  v8bf hi = *(const v8bf*)(row + kb + 16);
  return __builtin_shufflevector(lo, hi, 0, 1, 2, 3, 4, 5, 6, 7,
                                         8, 9, 10, 11, 12, 13, 14, 15);
}

// A fragment from a global f32 row pointer (converted to bf16); gx only.
__device__ __forceinline__ v16bf load_afrag_g(const float* row, int kb /*hi*8*/) {
  v16bf r;
#pragma unroll
  for (int e = 0; e < 8; ++e) {
    r[e]     = (__bf16)row[kb + e];
    r[e + 8] = (__bf16)row[kb + 16 + e];
  }
  return r;
}

// B fragment (32x16, bf16): lane = N = column; K = kb+e (e<8) / kb+16+(e-8).
// Two aligned 16B vector loads.
__device__ __forceinline__ v16bf load_bfrag(const __bf16* p, int kb /*hi*8*/) {
  v8bf lo = *(const v8bf*)(p + kb);
  v8bf hi = *(const v8bf*)(p + kb + 16);
  return __builtin_shufflevector(lo, hi, 0, 1, 2, 3, 4, 5, 6, 7,
                                         8, 9, 10, 11, 12, 13, 14, 15);
}

// ---------------------------------------------------------------------------
// One-time f32 -> bf16 weight conversion
// ---------------------------------------------------------------------------
__global__ void cvt_bf16_kernel(const float* __restrict__ src,
                                __bf16* __restrict__ dst, int n) {
  int i = blockIdx.x * 256 + threadIdx.x;
  if (i < n) dst[i] = (__bf16)src[i];
}

// ---------------------------------------------------------------------------
// Copy root hiddens (node 0 rows) into allh[b*H + h]   (8192 floats)
// ---------------------------------------------------------------------------
__global__ void root_copy_kernel(const float* __restrict__ root,
                                 float* __restrict__ allh) {
  int idx = blockIdx.x * 256 + threadIdx.x;
  allh[idx] = root[idx];
}

// ---------------------------------------------------------------------------
// gx[b, col] = sum_h ann_table[ann_ids[b]][h] * w_ih[col][h] + b_ih[col]
// single-wave tiles: grid (2 row tiles, 144 col tiles)
// ---------------------------------------------------------------------------
__global__ __launch_bounds__(32)
void gx_kernel(const float* __restrict__ ann_table, const int* __restrict__ ann_ids,
               const __bf16* __restrict__ w_ih, const float* __restrict__ b_ih,
               float* __restrict__ gx) {
  const int lane = threadIdx.x, lane15 = lane & 15, hi = lane >> 4;
  const int rt  = blockIdx.x;
  const int col = blockIdx.y * 16 + lane15;
  const float*  arow = ann_table + (size_t)ann_ids[rt * 16 + lane15] * Hc;
  const __bf16* wr   = w_ih + (size_t)col * Hc;
  v8f acc = {};
#pragma unroll
  for (int k0 = 0; k0 < Hc; k0 += 32)
    acc = wmma_bf16(load_afrag_g(arow + k0, hi * 8),
                    load_bfrag(wr + k0, hi * 8), acc);
  const float bb = b_ih[col];
#pragma unroll
  for (int i = 0; i < 8; ++i) {
    int b = rt * 16 + i + hi * 8;
    gx[(size_t)b * AGc + col] = acc[i] + bb;
  }
}

// ---------------------------------------------------------------------------
// One GRU level. One block per parent node (32 batch rows).
// 8 waves x 6 (a, h-block) units; each unit: 3 gates x 2 row-tiles.
// Parent activations staged in LDS as bf16, shared by all waves.
// ---------------------------------------------------------------------------
__global__ __launch_bounds__(256)
void gru_level_kernel(float* allh, const float* __restrict__ gx,
                      const __bf16* __restrict__ w_hh, const float* __restrict__ b_hh,
                      int parentStart, int childStart) {
  __shared__ __bf16 ah[32][280];   // 32 x 256, padded (row stride 560B, 16B-mult)
  const int tid  = threadIdx.x;
  const int p    = blockIdx.x;
  const int node = parentStart + p;
  const int gr0  = node * Bc;      // global row base for this node

  {  // stage 32x256 parent tile into LDS (bf16)
    int row = tid >> 3;
    int c0  = (tid & 7) * 32;
    const float* src = allh + (size_t)(gr0 + row) * Hc + c0;
#pragma unroll
    for (int e = 0; e < 32; ++e) ah[row][c0 + e] = (__bf16)src[e];
  }
  __syncthreads();

  const int wave = tid >> 5, lane = tid & 31, lane15 = lane & 15, hi = lane >> 4;

  for (int u = 0; u < 6; ++u) {
    const int c  = wave * 6 + u;         // 0..47
    const int a  = c >> 4;               // arity index
    const int hb = (c & 15) * 16;        // h block
    const int h  = hb + lane15;
    const __bf16* wr = w_hh + ((size_t)a * Gc + h) * Hc;         // r row
    const __bf16* wz = wr + (size_t)256 * Hc;                    // z row
    const __bf16* wn = wr + (size_t)512 * Hc;                    // n row

    v8f acc[6] = {};  // [gate*2 + rowtile]: r0 r1 z0 z1 n0 n1
#pragma unroll
    for (int k0 = 0; k0 < Hc; k0 += 32) {
      v16bf a0 = load_afrag_lds(&ah[lane15][k0],      hi * 8);
      v16bf a1 = load_afrag_lds(&ah[16 + lane15][k0], hi * 8);
      v16bf br = load_bfrag(wr + k0, hi * 8);
      v16bf bz = load_bfrag(wz + k0, hi * 8);
      v16bf bn = load_bfrag(wn + k0, hi * 8);
      acc[0] = wmma_bf16(a0, br, acc[0]);
      acc[1] = wmma_bf16(a1, br, acc[1]);
      acc[2] = wmma_bf16(a0, bz, acc[2]);
      acc[3] = wmma_bf16(a1, bz, acc[3]);
      acc[4] = wmma_bf16(a0, bn, acc[4]);
      acc[5] = wmma_bf16(a1, bn, acc[5]);
    }

    const float bhr = b_hh[a * Gc + h];
    const float bhz = b_hh[a * Gc + 256 + h];
    const float bhn = b_hh[a * Gc + 512 + h];
    const size_t childRow0 = (size_t)(childStart + p * 3 + a) * Bc;

#pragma unroll
    for (int rt = 0; rt < 2; ++rt) {
#pragma unroll
      for (int i = 0; i < 8; ++i) {
        const int b = rt * 16 + i + hi * 8;      // batch index 0..31
        const float* gxr = gx + (size_t)b * AGc + a * Gc;
        float r  = sigmoidf_(gxr[h]       + acc[0 + rt][i] + bhr);
        float z  = sigmoidf_(gxr[256 + h] + acc[2 + rt][i] + bhz);
        float nn = tanhf   (gxr[512 + h] + r * (acc[4 + rt][i] + bhn));
        float hprev = allh[(size_t)(gr0 + b) * Hc + h];
        allh[(childRow0 + b) * Hc + h] = (1.0f - z) * nn + z * hprev;
      }
    }
  }
}

// ---------------------------------------------------------------------------
// Head: one block per node. Stage A in LDS, mid = sigmoid(A*w1^T+b1) kept in
// LDS as bf16 (32x512), then out = mid*w2^T+b2 written at pre-order position.
// ---------------------------------------------------------------------------
__global__ __launch_bounds__(256)
void head_kernel(const float* __restrict__ allh,
                 const __bf16* __restrict__ w1, const float* __restrict__ b1,
                 const __bf16* __restrict__ w2, const float* __restrict__ b2,
                 float* __restrict__ out) {
  __shared__ __bf16 ah[32][280];    // 32 x 256 activations
  __shared__ __bf16 mid[32][536];   // 32 x 512 intermediate (stride 1072B)
  const int tid  = threadIdx.x;
  const int node = blockIdx.x;
  const int gr0  = node * Bc;

  {  // stage activations
    int row = tid >> 3;
    int c0  = (tid & 7) * 32;
    const float* src = allh + (size_t)(gr0 + row) * Hc + c0;
#pragma unroll
    for (int e = 0; e < 32; ++e) ah[row][c0 + e] = (__bf16)src[e];
  }

  // pre-order position of BFS node (uniform across the block)
  int pos;
  {
    const int LS[9] = {0, 1, 4, 13, 40, 121, 364, 1093, 3280};
    const int T[8]  = {0, 1093, 364, 121, 40, 13, 4, 1};
    int l = 0;
    while (node >= LS[l + 1]) ++l;
    int pp = node - LS[l];
    int pw = 1;
    for (int i = 1; i < l; ++i) pw *= 3;
    pos = 0;
    for (int i = 1; i <= l; ++i) {
      int d = pp / pw; pp -= d * pw;
      pos += 1 + d * T[i];
      pw /= 3;
    }
  }
  __syncthreads();

  const int wave = tid >> 5, lane = tid & 31, lane15 = lane & 15, hi = lane >> 4;

  // ---- stage 1: mid = sigmoid(ah * w1^T + b1), K = 256 ----
  for (int u = 0; u < 4; ++u) {
    const int col = (wave * 4 + u) * 16 + lane15;   // 0..511
    const __bf16* wr = w1 + (size_t)col * Hc;
    v8f acc0 = {}, acc1 = {};
#pragma unroll
    for (int k0 = 0; k0 < Hc; k0 += 32) {
      v16bf a0 = load_afrag_lds(&ah[lane15][k0],      hi * 8);
      v16bf a1 = load_afrag_lds(&ah[16 + lane15][k0], hi * 8);
      v16bf bf = load_bfrag(wr + k0, hi * 8);
      acc0 = wmma_bf16(a0, bf, acc0);
      acc1 = wmma_bf16(a1, bf, acc1);
    }
    const float bb = b1[col];
#pragma unroll
    for (int i = 0; i < 8; ++i) {
      int m = i + hi * 8;
      mid[m][col]      = (__bf16)sigmoidf_(acc0[i] + bb);
      mid[16 + m][col] = (__bf16)sigmoidf_(acc1[i] + bb);
    }
  }
  __syncthreads();

  // ---- stage 2: out = mid * w2^T + b2, K = 512 ----
  const size_t outBase = (size_t)pos * (Bc * Vc);
  for (int u = 0; u < 4; ++u) {
    const int v = (wave * 4 + u) * 16 + lane15;     // 0..511
    const __bf16* wr = w2 + (size_t)v * Vc;
    v8f acc0 = {}, acc1 = {};
#pragma unroll
    for (int k0 = 0; k0 < Vc; k0 += 32) {
      v16bf a0 = load_afrag_lds(&mid[lane15][k0],      hi * 8);
      v16bf a1 = load_afrag_lds(&mid[16 + lane15][k0], hi * 8);
      v16bf bf = load_bfrag(wr + k0, hi * 8);
      acc0 = wmma_bf16(a0, bf, acc0);
      acc1 = wmma_bf16(a1, bf, acc1);
    }
    const float bb = b2[v];
#pragma unroll
    for (int i = 0; i < 8; ++i) {
      int m = i + hi * 8;
      out[outBase + (size_t)m * Vc + v]        = acc0[i] + bb;
      out[outBase + (size_t)(16 + m) * Vc + v] = acc1[i] + bb;
    }
  }
}

// ---------------------------------------------------------------------------
extern "C" void kernel_launch(void* const* d_in, const int* in_sizes, int n_in,
                              void* d_out, int out_size, void* d_ws, size_t ws_size,
                              hipStream_t stream) {
  (void)in_sizes; (void)n_in; (void)out_size; (void)ws_size;
  const float* root      = (const float*)d_in[0];
  const float* ann_table = (const float*)d_in[1];
  const float* w1        = (const float*)d_in[2];
  const float* b1        = (const float*)d_in[3];
  const float* w2        = (const float*)d_in[4];
  const float* b2        = (const float*)d_in[5];
  const float* w_ih      = (const float*)d_in[6];
  const float* w_hh      = (const float*)d_in[7];
  const float* b_ih      = (const float*)d_in[8];
  const float* b_hh      = (const float*)d_in[9];
  const int*   ann_ids   = (const int*)d_in[10];

  // f32 workspace region
  float* gx   = (float*)d_ws;          // 32 * 2304 floats
  float* allh = gx + Bc * AGc;         // 3280 * 32 * 256 floats (~107.5 MB)
  // bf16 workspace region (after allh)
  __bf16* w_ih_bf = (__bf16*)(allh + (size_t)NNODES * Bc * Hc);
  __bf16* w_hh_bf = w_ih_bf + (size_t)AGc * Hc;     // 589824
  __bf16* w1_bf   = w_hh_bf + (size_t)AGc * Hc;     // 589824
  __bf16* w2_bf   = w1_bf + (size_t)Vc * Hc;        // 131072
  //                w2_bf: 262144 bf16

  const int n_ih = AGc * Hc, n_hh = AGc * Hc, n_w1 = Vc * Hc, n_w2 = Vc * Vc;
  cvt_bf16_kernel<<<(n_ih + 255) / 256, 256, 0, stream>>>(w_ih, w_ih_bf, n_ih);
  cvt_bf16_kernel<<<(n_hh + 255) / 256, 256, 0, stream>>>(w_hh, w_hh_bf, n_hh);
  cvt_bf16_kernel<<<(n_w1 + 255) / 256, 256, 0, stream>>>(w1, w1_bf, n_w1);
  cvt_bf16_kernel<<<(n_w2 + 255) / 256, 256, 0, stream>>>(w2, w2_bf, n_w2);

  root_copy_kernel<<<32, 256, 0, stream>>>(root, allh);
  gx_kernel<<<dim3(2, AGc / 16), 32, 0, stream>>>(ann_table, ann_ids, w_ih_bf, b_ih, gx);

  const int LS[8] = {0, 1, 4, 13, 40, 121, 364, 1093};
  int nl = 1;
  for (int l = 0; l < 7; ++l) {
    gru_level_kernel<<<nl, 256, 0, stream>>>(allh, gx, w_hh_bf, b_hh, LS[l], LS[l + 1]);
    nl *= 3;
  }

  head_kernel<<<NNODES, 256, 0, stream>>>(allh, w1_bf, b1, w2_bf, b2, (float*)d_out);
}
